// LlamaAttention_33861522162626
// MI455X (gfx1250) — matrix operand order repaired and checked by reading, not compiled, not run on hardware
//
#include <hip/hip_runtime.h>
#include <stdint.h>

// ---------------------------------------------------------------------------
// CDNA5 (gfx1250) WMMA / TDM types
// ---------------------------------------------------------------------------
typedef __attribute__((ext_vector_type(16))) __bf16       v16bf;
typedef __attribute__((ext_vector_type(8)))  float        v8f;
typedef __attribute__((ext_vector_type(4)))  unsigned int v4u;
typedef __attribute__((ext_vector_type(8)))  int          v8i;
typedef __attribute__((ext_vector_type(4)))  int          v4i;

#if defined(__has_builtin)
#if __has_builtin(__builtin_amdgcn_tensor_load_to_lds)
#define HAVE_TDM 1
#endif
#endif
#ifndef HAVE_TDM
#define HAVE_TDM 0
#endif

#define WMMA_BF16(a, b, c)                                                     \
  __builtin_amdgcn_wmma_f32_16x16x32_bf16(false, (a), false, (b), (short)0,    \
                                          (c), false, false)

// A-fragment (16x32 bf16, M x K), contiguous k order: k = 16*(lane>>4) + i
__device__ __forceinline__ v16bf ld_frag_a(const __bf16* __restrict__ base,
                                           int ld) {
  const int lane = threadIdx.x & 31;
  const int r  = lane & 15;
  const int kb = (lane >> 4) * 16;
  const __bf16* p = base + r * ld + kb;
  v16bf f;
#pragma unroll
  for (int i = 0; i < 16; ++i) f[i] = p[i];
  return f;
}

// B-fragment (32x16 bf16, K x N), memory layout N x K row-major
__device__ __forceinline__ v16bf ld_frag_b(const __bf16* __restrict__ base,
                                           int ld) {
  const int lane = threadIdx.x & 31;
  const int n  = lane & 15;
  const int kb = (lane >> 4) * 16;
  const __bf16* p = base + n * ld + kb;
  v16bf f;
#pragma unroll
  for (int i = 0; i < 16; ++i) f[i] = p[i];
  return f;
}

// ---------------------------------------------------------------------------
// TDM: async 2-D tile load (128 rows x 32 bf16) global -> LDS.
// HW pad: every 16 DWORDs (one 64B row) insert 4 DWORDs -> LDS row stride of
// 40 bf16 (80B): 16B-aligned for ds_load_b128, conflict-free bank pattern.
// ---------------------------------------------------------------------------
constexpr int BM  = 128;
constexpr int BN  = 128;
constexpr int BK  = 32;
constexpr int LDT = 40;  // LDS row stride in elements (matches TDM pad above)

#if HAVE_TDM
__device__ __forceinline__ void tdm_load_tile_128x32_bf16(
    const void* gptr, uint32_t lds_byte, int row_stride_elems) {
  const uint64_t ga = (uint64_t)(uintptr_t)gptr;
  v4u g0;
  g0[0] = 1u;                                    // count = 1 (valid descriptor)
  g0[1] = lds_byte;                              // LDS byte address
  g0[2] = (uint32_t)ga;                          // global_addr[31:0]
  g0[3] = (uint32_t)((ga >> 32) & 0x01FFFFFFu)   // global_addr[56:32]
        | 0x80000000u;                           // type = 2 ("image")
  v8i g1;
  g1[0] = (int)((1u << 16)     // data_size = 2 bytes
              | (1u << 20)     // pad_enable
              | (3u << 22)     // pad_interval: 16 DWORDs (= one 32-elem row)
              | (3u << 25));   // pad_amount: 4 DWORDs (= 8 bf16)
  g1[1] = (int)(32u << 16);    // tensor_dim0 = 32 (tile-sized; never OOB)
  g1[2] = (int)(128u << 16);   // tensor_dim1 = 128
  g1[3] = (int)(32u << 16);    // tile_dim0 = 32
  g1[4] = (int)128u;           // tile_dim1 = 128
  g1[5] = row_stride_elems;    // tensor_dim0_stride[31:0] (elements)
  g1[6] = 0;
  g1[7] = 0;
  const v4i z4 = {0, 0, 0, 0};
#if defined(__clang_major__) && (__clang_major__ >= 23)
  const v8i z8 = {0, 0, 0, 0, 0, 0, 0, 0};
  __builtin_amdgcn_tensor_load_to_lds(g0, g1, z4, z4, z8, 0);
#else
  __builtin_amdgcn_tensor_load_to_lds(g0, g1, z4, z4, 0);
#endif
}
#endif

// ---------------------------------------------------------------------------
// fp32 -> bf16 streaming conversion (one-time; makes GEMM set L2-resident)
// ---------------------------------------------------------------------------
__global__ __launch_bounds__(256)
void f32_to_bf16_kernel(const float* __restrict__ src, __bf16* __restrict__ dst,
                        long n) {
  const long i = ((long)blockIdx.x * blockDim.x + threadIdx.x) * 8;
  if (i + 8 <= n) {
#pragma unroll
    for (int j = 0; j < 8; ++j) dst[i + j] = (__bf16)src[i + j];
  }
}

// VIB masks: sigmoid(mu + eps * sigma), length 64 each
__global__ void vib_mask_kernel(const float* __restrict__ mu_q,
                                const float* __restrict__ sg_q,
                                const float* __restrict__ ep_q,
                                const float* __restrict__ mu_kv,
                                const float* __restrict__ sg_kv,
                                const float* __restrict__ ep_kv,
                                float* __restrict__ mq,
                                float* __restrict__ mkv) {
  const int i = threadIdx.x;
  if (i < 64) {
    mq[i]  = 1.0f / (1.0f + __expf(-(mu_q[i]  + ep_q[i]  * sg_q[i])));
    mkv[i] = 1.0f / (1.0f + __expf(-(mu_kv[i] + ep_kv[i] * sg_kv[i])));
  }
}

// ---------------------------------------------------------------------------
// GEMM (TN): C[M,N] = A[M,K] * B[N,K]^T, both operands bf16, fp32 accumulate.
// 128x128 block tile, BK=32, 8 waves (wave tile 32x64), TDM double-buffered
// LDS staging (wave 0 issues; 1 barrier per K-step). Optional fused per-column
// VIB scale colmask[col & 63]; output bf16 (workspace) or fp32 (d_out).
// ---------------------------------------------------------------------------
template <bool OUT_BF16>
__global__ __launch_bounds__(256)
void gemm_tn_bf16(const __bf16* __restrict__ A, const __bf16* __restrict__ B,
                  void* __restrict__ Cout, const float* __restrict__ colmask,
                  const int Kdim, const int Ndim) {
  __shared__ __align__(16) __bf16 sA[2][BM * LDT];
  __shared__ __align__(16) __bf16 sB[2][BN * LDT];

  const int tid  = threadIdx.x;
  const int lane = tid & 31;
  const int wave = tid >> 5;
  const int wr   = wave >> 1;  // 0..3 : 32-row strip
  const int wc   = wave & 1;   // 0..1 : 64-col strip
  const long block_m = (long)blockIdx.x * BM;
  const long block_n = (long)blockIdx.y * BN;

  const __bf16* Ab = A + block_m * (long)Kdim;
  const __bf16* Bb = B + block_n * (long)Kdim;
  const int iters = Kdim / BK;

  v8f acc[2][4] = {};

#if HAVE_TDM
  const uint32_t ldsA[2] = {(uint32_t)(uintptr_t)&sA[0][0],
                            (uint32_t)(uintptr_t)&sA[1][0]};
  const uint32_t ldsB[2] = {(uint32_t)(uintptr_t)&sB[0][0],
                            (uint32_t)(uintptr_t)&sB[1][0]};
  if (tid < 32) {  // wave 0 drives the DMA (TDM ignores EXEC; branch is per-wave)
    tdm_load_tile_128x32_bf16(Ab, ldsA[0], Kdim);
    tdm_load_tile_128x32_bf16(Bb, ldsB[0], Kdim);
  }
#else
  const int lrow = tid >> 1;
  const int lcol = (tid & 1) * 16;
#endif

  for (int i = 0; i < iters; ++i) {
    const int cur = i & 1;
#if HAVE_TDM
    if (tid < 32) __builtin_amdgcn_s_wait_tensorcnt(0);
    __syncthreads();
    asm volatile("" ::: "memory");  // TDM wrote LDS invisibly to alias analysis
    if ((tid < 32) && (i + 1 < iters)) {
      const long k1 = (long)(i + 1) * BK;
      tdm_load_tile_128x32_bf16(Ab + k1, ldsA[cur ^ 1], Kdim);
      tdm_load_tile_128x32_bf16(Bb + k1, ldsB[cur ^ 1], Kdim);
    }
#else
    {
      const long k0 = (long)i * BK;
      const __bf16* ga = Ab + (long)lrow * Kdim + k0 + lcol;
      const __bf16* gb = Bb + (long)lrow * Kdim + k0 + lcol;
      __bf16* da = &sA[cur][lrow * LDT + lcol];
      __bf16* db = &sB[cur][lrow * LDT + lcol];
#pragma unroll
      for (int j = 0; j < 16; ++j) da[j] = ga[j];
#pragma unroll
      for (int j = 0; j < 16; ++j) db[j] = gb[j];
    }
    __syncthreads();  // double buffer -> one barrier per K-step suffices
#endif

    const __bf16* tA = &sA[cur][0];
    const __bf16* tB = &sB[cur][0];
    const v16bf a0 = ld_frag_a(tA + (wr * 32 + 0)  * LDT, LDT);
    const v16bf a1 = ld_frag_a(tA + (wr * 32 + 16) * LDT, LDT);
    const v16bf b0 = ld_frag_b(tB + (wc * 64 + 0)  * LDT, LDT);
    const v16bf b1 = ld_frag_b(tB + (wc * 64 + 16) * LDT, LDT);
    const v16bf b2 = ld_frag_b(tB + (wc * 64 + 32) * LDT, LDT);
    const v16bf b3 = ld_frag_b(tB + (wc * 64 + 48) * LDT, LDT);

    acc[0][0] = WMMA_BF16(a0, b0, acc[0][0]);
    acc[0][1] = WMMA_BF16(a0, b1, acc[0][1]);
    acc[0][2] = WMMA_BF16(a0, b2, acc[0][2]);
    acc[0][3] = WMMA_BF16(a0, b3, acc[0][3]);
    acc[1][0] = WMMA_BF16(a1, b0, acc[1][0]);
    acc[1][1] = WMMA_BF16(a1, b1, acc[1][1]);
    acc[1][2] = WMMA_BF16(a1, b2, acc[1][2]);
    acc[1][3] = WMMA_BF16(a1, b3, acc[1][3]);
  }

  // Epilogue. C layout: VGPR i of a 16x16 tile = (row = i + 8*(lane>>4), col = lane&15)
#pragma unroll
  for (int r = 0; r < 2; ++r) {
#pragma unroll
    for (int t = 0; t < 4; ++t) {
      const long gc = block_n + wc * 64 + t * 16 + (lane & 15);
      const float scale = colmask ? colmask[gc & 63] : 1.0f;
      const long gr0 = block_m + wr * 32 + r * 16 + (lane >> 4) * 8;
#pragma unroll
      for (int i = 0; i < 8; ++i) {
        const float v = acc[r][t][i] * scale;
        if constexpr (OUT_BF16)
          ((__bf16*)Cout)[(gr0 + i) * (long)Ndim + gc] = (__bf16)v;
        else
          ((float*)Cout)[(gr0 + i) * (long)Ndim + gc] = v;
      }
    }
  }
}

// ---------------------------------------------------------------------------
// Per-token cross-head attention: S = (Q K^T)/8 ; P = softmax(S) ; O = P V
// (all 64x64 per token). One workgroup (4 waves) per token; O aliases Q.
// ---------------------------------------------------------------------------
__global__ __launch_bounds__(128)
void attn_per_token_kernel(const __bf16* __restrict__ Q,
                           const __bf16* __restrict__ Kmat,
                           const __bf16* __restrict__ V,
                           __bf16* __restrict__ O) {
  constexpr int LS = 72;  // 144B row stride: 16B-aligned, low bank conflict
  __shared__ __align__(16) __bf16 sQ [64 * LS];
  __shared__ __align__(16) __bf16 sK [64 * LS];
  __shared__ __align__(16) __bf16 sVt[64 * LS];
  __shared__ __align__(16) __bf16 sP [64 * LS];

  const int tid  = threadIdx.x;
  const int lane = tid & 31;
  const int wave = tid >> 5;  // 0..3, each owns a 16-row strip
  const size_t base = (size_t)blockIdx.x * 4096;

  {
    const int r  = tid >> 1;
    const int c0 = (tid & 1) * 32;
#pragma unroll 8
    for (int i = 0; i < 32; ++i) {
      const int c = c0 + i;
      sQ[r * LS + c]  = Q[base + r * 64 + c];
      sK[r * LS + c]  = Kmat[base + r * 64 + c];
      sVt[c * LS + r] = V[base + r * 64 + c];  // transpose: PV needs B = V^T
    }
  }
  __syncthreads();

  v8f acc[4] = {};
#pragma unroll
  for (int kk = 0; kk < 64; kk += 32) {
    const v16bf a = ld_frag_a(sQ + wave * 16 * LS + kk, LS);
#pragma unroll
    for (int t = 0; t < 4; ++t) {
      const v16bf b = ld_frag_b(sK + t * 16 * LS + kk, LS);
      acc[t] = WMMA_BF16(a, b, acc[t]);
    }
  }

  // Row softmax: register row i lives in one 16-lane half-group, so
  // xor-shuffles 1/2/4/8 reduce each row independently.
#pragma unroll
  for (int i = 0; i < 8; ++i) {
    float x[4], m = -3.0e38f;
#pragma unroll
    for (int t = 0; t < 4; ++t) { x[t] = acc[t][i] * 0.125f; m = fmaxf(m, x[t]); }
#pragma unroll
    for (int s = 1; s < 16; s <<= 1) m = fmaxf(m, __shfl_xor(m, s, 32));
    float sum = 0.0f;
#pragma unroll
    for (int t = 0; t < 4; ++t) { x[t] = __expf(x[t] - m); sum += x[t]; }
#pragma unroll
    for (int s = 1; s < 16; s <<= 1) sum += __shfl_xor(sum, s, 32);
    const float inv = 1.0f / sum;
#pragma unroll
    for (int t = 0; t < 4; ++t) {
      const int r = wave * 16 + (lane >> 4) * 8 + i;
      const int c = t * 16 + (lane & 15);
      sP[r * LS + c] = (__bf16)(x[t] * inv);
    }
  }
  __syncthreads();

  v8f o[4] = {};
#pragma unroll
  for (int kk = 0; kk < 64; kk += 32) {
    const v16bf a = ld_frag_a(sP + wave * 16 * LS + kk, LS);
#pragma unroll
    for (int t = 0; t < 4; ++t) {
      const v16bf b = ld_frag_b(sVt + t * 16 * LS + kk, LS);
      o[t] = WMMA_BF16(a, b, o[t]);
    }
  }

#pragma unroll
  for (int t = 0; t < 4; ++t) {
#pragma unroll
    for (int i = 0; i < 8; ++i) {
      const int h = wave * 16 + (lane >> 4) * 8 + i;
      const int d = t * 16 + (lane & 15);
      O[base + h * 64 + d] = (__bf16)o[t][i];
    }
  }
}

// ---------------------------------------------------------------------------
// Launch: bf16 converts -> masks -> Q/K/V GEMMs (fused VIB mask, bf16 out) ->
// per-token attention (out aliases Q) -> output projection (fp32 to d_out).
// Weight slot (32MB) converted lazily and reused for all four weights.
// ---------------------------------------------------------------------------
extern "C" void kernel_launch(void* const* d_in, const int* in_sizes, int n_in,
                              void* d_out, int out_size, void* d_ws,
                              size_t ws_size, hipStream_t stream) {
  (void)in_sizes; (void)n_in; (void)out_size; (void)ws_size;

  const float* hs    = (const float*)d_in[0];
  const float* Wq    = (const float*)d_in[1];
  const float* Wk    = (const float*)d_in[2];
  const float* Wv    = (const float*)d_in[3];
  const float* Wo    = (const float*)d_in[4];
  const float* mu_q  = (const float*)d_in[5];
  const float* sg_q  = (const float*)d_in[6];
  const float* mu_kv = (const float*)d_in[7];
  const float* sg_kv = (const float*)d_in[8];
  const float* ep_q  = (const float*)d_in[9];
  const float* ep_kv = (const float*)d_in[10];

  const int  Mtok  = 2 * 4096;            // B*S tokens
  const int  Kd    = 4096;
  const int  Nd    = 4096;
  const long matMK = (long)Mtok * Kd;     // activation matrix elements
  const long matNK = (long)Nd * Kd;       // weight matrix elements

  // Workspace (bf16): hs | weight-slot | q | k | v | masks   (~288 MB)
  __bf16* hsb   = (__bf16*)d_ws;
  __bf16* wslot = hsb + matMK;
  __bf16* qb    = wslot + matNK;
  __bf16* kb    = qb + matMK;
  __bf16* vb    = kb + matMK;
  float*  mq    = (float*)(vb + matMK);
  float*  mkv   = mq + 64;

  const int CT = 256;
  f32_to_bf16_kernel<<<(int)(matMK / (8 * CT)), CT, 0, stream>>>(hs, hsb, matMK);
  vib_mask_kernel<<<1, 64, 0, stream>>>(mu_q, sg_q, ep_q, mu_kv, sg_kv, ep_kv,
                                        mq, mkv);

  dim3 grid(Mtok / BM, Nd / BN);
  f32_to_bf16_kernel<<<(int)(matNK / (8 * CT)), CT, 0, stream>>>(Wq, wslot, matNK);
  gemm_tn_bf16<true><<<grid, 256, 0, stream>>>(hsb, wslot, qb, mq, Kd, Nd);
  f32_to_bf16_kernel<<<(int)(matNK / (8 * CT)), CT, 0, stream>>>(Wk, wslot, matNK);
  gemm_tn_bf16<true><<<grid, 256, 0, stream>>>(hsb, wslot, kb, mkv, Kd, Nd);
  f32_to_bf16_kernel<<<(int)(matNK / (8 * CT)), CT, 0, stream>>>(Wv, wslot, matNK);
  gemm_tn_bf16<true><<<grid, 256, 0, stream>>>(hsb, wslot, vb, mkv, Kd, Nd);

  attn_per_token_kernel<<<Mtok, 128, 0, stream>>>(qb, kb, vb, /*O aliases*/ qb);

  f32_to_bf16_kernel<<<(int)(matNK / (8 * CT)), CT, 0, stream>>>(Wo, wslot, matNK);
  gemm_tn_bf16<false><<<grid, 256, 0, stream>>>(qb, wslot, d_out, nullptr, Kd, Nd);
}
